// Recurrent_Neural_Network_26233660244792
// MI455X (gfx1250) — matrix operand order repaired and checked by reading, not compiled, not run on hardware
//
#include <hip/hip_runtime.h>
#include <hip/hip_bf16.h>
#include <math.h>

#define IN_DIM  512
#define H_DIM   1024
#define B_DIM   256
#define T_DIM   512
#define OUT_DIM 512

// Padded LDS row strides (elements). Row stride in dwords % 64 banks == 4,
// so 16 fragment rows map to 16*4 = 64 distinct banks (conflict-free b128),
// and row stride stays a multiple of 16 bytes (520*2=1040, 1032*2=2064).
#define LDA_X (IN_DIM + 8)   // 520
#define LDA_H (H_DIM + 8)    // 1032

typedef __attribute__((ext_vector_type(16))) __bf16 v16bf;
typedef __attribute__((ext_vector_type(8)))  float  v8f;

union FragBF { v16bf v; uint4 q[2]; };

// ---------------------------------------------------------------------------
// Fragment loaders (wave32, gfx1250 WMMA layouts from CDNA5 ISA)
// ---------------------------------------------------------------------------

// A-matrix 16x32 bf16 from LDS tile (rows 0..15, padded stride ld):
// lane<16 -> row=lane,    K = [k0..k0+7] and [k0+16..k0+23]
// lane>=16 -> row=lane-16, K = [k0+8..k0+15] and [k0+24..k0+31]
__device__ __forceinline__ v16bf load_a_lds(const __bf16* S, int ld, int k0, int lane) {
  const int sel = lane >> 4;
  const int row = lane & 15;
  const uint4* p = (const uint4*)(S + row * ld + k0 + sel * 8);
  FragBF f;
  f.q[0] = p[0];   // 8 bf16: K = k0 + sel*8 .. +7
  f.q[1] = p[2];   // 8 bf16: K = k0 + 16 + sel*8 .. +7
  return f.v;
}

// A-matrix from global bf16 row-major (used by FC kernel only).
__device__ __forceinline__ v16bf load_a_bf16(const __bf16* __restrict__ A, int ld,
                                             int m0, int k0, int lane) {
  const int sel = lane >> 4;
  const int row = m0 + (lane & 15);
  const uint4* p = (const uint4*)(A + (size_t)row * ld + k0 + sel * 8);
  FragBF f;
  f.q[0] = p[0];
  f.q[1] = p[2];
  return f.v;
}

// B-matrix 32x16 bf16 for C = A * W^T:  B[k][n] = W[n][k].
// lane<16 -> col=lane,     K = [k0..k0+7] and [k0+16..k0+23]
// lane>=16 -> col=lane-16, K = [k0+8..k0+15] and [k0+24..k0+31]
__device__ __forceinline__ v16bf load_b_bf16(const __bf16* __restrict__ W, int ld,
                                             int n0, int k0, int lane) {
  const int sel = lane >> 4;
  const int col = n0 + (lane & 15);
  const uint4* p = (const uint4*)(W + (size_t)col * ld + k0 + sel * 8);
  FragBF f;
  f.q[0] = p[0];
  f.q[1] = p[2];
  return f.v;
}

// ---------------------------------------------------------------------------
// Utility kernels
// ---------------------------------------------------------------------------
__global__ void cvt_f32_to_bf16(const float* __restrict__ src,
                                __bf16* __restrict__ dst, int n) {
  int i = blockIdx.x * blockDim.x + threadIdx.x;
  if (i < n) dst[i] = (__bf16)src[i];
}

__global__ void zero_bf16(__bf16* __restrict__ dst, int n) {
  int i = blockIdx.x * blockDim.x + threadIdx.x;
  if (i < n) dst[i] = (__bf16)0.0f;
}

// ---------------------------------------------------------------------------
// LDS staging helpers (per block: one 16-row A tile shared by all 8 waves)
// ---------------------------------------------------------------------------

// Stage 16 x 1024 bf16 rows from global (row stride H_DIM) into padded LDS.
__device__ __forceinline__ void stage_h_tile(const __bf16* __restrict__ src, int bm0,
                                             __bf16* S, int tid) {
#pragma unroll
  for (int it = 0; it < 8; ++it) {           // 16*1024 elems / (256 thr * 8)
    const int e = (it * 256 + tid) * 8;
    const int r = e >> 10;
    const int c = e & 1023;
    const uint4 v = *(const uint4*)(src + (size_t)(bm0 + r) * H_DIM + c);
    *(uint4*)&S[r * LDA_H + c] = v;
  }
}

// Stage 16 x 512 fp32 rows of x[:, t, :] into padded bf16 LDS (convert once).
__device__ __forceinline__ void stage_x_tile(const float* __restrict__ x, int t, int bm0,
                                             __bf16* S, int tid) {
#pragma unroll
  for (int it = 0; it < 4; ++it) {           // 16*512 elems / (256 thr * 8)
    const int e = (it * 256 + tid) * 8;
    const int r = e >> 9;
    const int c = e & 511;
    const float4* src =
        (const float4*)(x + ((size_t)(bm0 + r) * T_DIM + t) * IN_DIM + c);
    const float4 a = src[0];
    const float4 b = src[1];
    union { __bf16 h[8]; uint4 q; } u;
    u.h[0] = (__bf16)a.x; u.h[1] = (__bf16)a.y; u.h[2] = (__bf16)a.z; u.h[3] = (__bf16)a.w;
    u.h[4] = (__bf16)b.x; u.h[5] = (__bf16)b.y; u.h[6] = (__bf16)b.z; u.h[7] = (__bf16)b.w;
    *(uint4*)&S[r * LDA_X + c] = u.q;
  }
}

// ---------------------------------------------------------------------------
// Layer 0 step: h0_next = tanh(x_t @ Wih0^T + h0_prev @ Whh0^T + b_ih + b_hh)
// grid = 128 blocks * 256 thr = 1024 waves == (B/16)*(H/16) tiles.
// Blocks [8g..8g+7] share batch-row group g; each wave owns one 16x16 tile.
// ---------------------------------------------------------------------------
__global__ __launch_bounds__(256) void rnn_step_l0(
    const float* __restrict__ x, int t,
    const __bf16* __restrict__ h_prev,
    const __bf16* __restrict__ w_ih, const __bf16* __restrict__ w_hh,
    const float* __restrict__ b_ih, const float* __restrict__ b_hh,
    __bf16* __restrict__ h_next) {
  __shared__ __bf16 sX[16 * LDA_X];
  __shared__ __bf16 sH[16 * LDA_H];

  const int tid  = threadIdx.x;
  const int lane = tid & 31;
  const int wid  = tid >> 5;
  const int bm0  = (blockIdx.x >> 3) << 4;                 // batch tile origin
  const int n0   = (((blockIdx.x & 7) << 3) | wid) << 4;   // hidden tile origin

  stage_x_tile(x, t, bm0, sX, tid);
  stage_h_tile(h_prev, bm0, sH, tid);
  __syncthreads();

  v8f acc = {0.f, 0.f, 0.f, 0.f, 0.f, 0.f, 0.f, 0.f};

  // Input projection part: K = IN
#pragma unroll 4
  for (int k0 = 0; k0 < IN_DIM; k0 += 32) {
    __builtin_prefetch(w_ih + (size_t)(n0 + (lane & 15)) * IN_DIM + k0 + 128, 0, 3);
    v16bf a = load_a_lds(sX, LDA_X, k0, lane);
    v16bf b = load_b_bf16(w_ih, IN_DIM, n0, k0, lane);
    acc = __builtin_amdgcn_wmma_f32_16x16x32_bf16(false, a, false, b,
                                                  (short)0, acc, false, false);
  }
  // Recurrent part: K = H
#pragma unroll 4
  for (int k0 = 0; k0 < H_DIM; k0 += 32) {
    __builtin_prefetch(w_hh + (size_t)(n0 + (lane & 15)) * H_DIM + k0 + 128, 0, 3);
    v16bf a = load_a_lds(sH, LDA_H, k0, lane);
    v16bf b = load_b_bf16(w_hh, H_DIM, n0, k0, lane);
    acc = __builtin_amdgcn_wmma_f32_16x16x32_bf16(false, a, false, b,
                                                  (short)0, acc, false, false);
  }

  const int l = lane & 15, hi = lane >> 4;
  const int col = n0 + l;
  const float bs = b_ih[col] + b_hh[col];
#pragma unroll
  for (int v = 0; v < 8; ++v) {
    const int row = bm0 + v + (hi << 3);
    h_next[(size_t)row * H_DIM + col] = (__bf16)tanhf(acc[v] + bs);
  }
}

// ---------------------------------------------------------------------------
// Layer 1 step: h1_next = tanh(h0 @ Wih1^T + h1_prev @ Whh1^T + b_ih + b_hh)
// ---------------------------------------------------------------------------
__global__ __launch_bounds__(256) void rnn_step_l1(
    const __bf16* __restrict__ h0,
    const __bf16* __restrict__ h_prev,
    const __bf16* __restrict__ w_ih, const __bf16* __restrict__ w_hh,
    const float* __restrict__ b_ih, const float* __restrict__ b_hh,
    __bf16* __restrict__ h_next) {
  __shared__ __bf16 sH0[16 * LDA_H];
  __shared__ __bf16 sH1[16 * LDA_H];

  const int tid  = threadIdx.x;
  const int lane = tid & 31;
  const int wid  = tid >> 5;
  const int bm0  = (blockIdx.x >> 3) << 4;
  const int n0   = (((blockIdx.x & 7) << 3) | wid) << 4;

  stage_h_tile(h0, bm0, sH0, tid);
  stage_h_tile(h_prev, bm0, sH1, tid);
  __syncthreads();

  v8f acc = {0.f, 0.f, 0.f, 0.f, 0.f, 0.f, 0.f, 0.f};

#pragma unroll 4
  for (int k0 = 0; k0 < H_DIM; k0 += 32) {
    __builtin_prefetch(w_ih + (size_t)(n0 + (lane & 15)) * H_DIM + k0 + 128, 0, 3);
    v16bf a = load_a_lds(sH0, LDA_H, k0, lane);
    v16bf b = load_b_bf16(w_ih, H_DIM, n0, k0, lane);
    acc = __builtin_amdgcn_wmma_f32_16x16x32_bf16(false, a, false, b,
                                                  (short)0, acc, false, false);
  }
#pragma unroll 4
  for (int k0 = 0; k0 < H_DIM; k0 += 32) {
    __builtin_prefetch(w_hh + (size_t)(n0 + (lane & 15)) * H_DIM + k0 + 128, 0, 3);
    v16bf a = load_a_lds(sH1, LDA_H, k0, lane);
    v16bf b = load_b_bf16(w_hh, H_DIM, n0, k0, lane);
    acc = __builtin_amdgcn_wmma_f32_16x16x32_bf16(false, a, false, b,
                                                  (short)0, acc, false, false);
  }

  const int l = lane & 15, hi = lane >> 4;
  const int col = n0 + l;
  const float bs = b_ih[col] + b_hh[col];
#pragma unroll
  for (int v = 0; v < 8; ++v) {
    const int row = bm0 + v + (hi << 3);
    h_next[(size_t)row * H_DIM + col] = (__bf16)tanhf(acc[v] + bs);
  }
}

// ---------------------------------------------------------------------------
// FC: out = relu(h1 @ Wfc^T + b_fc), out fp32 [B, OUT]. Runs once.
// grid = 64 blocks * 256 thr = 512 waves == (B/16)*(OUT/16) tiles
// ---------------------------------------------------------------------------
__global__ __launch_bounds__(256) void fc_relu(
    const __bf16* __restrict__ h,
    const __bf16* __restrict__ w_fc,
    const float* __restrict__ b_fc,
    float* __restrict__ out) {
  const int lane = threadIdx.x & 31;
  const int wave = (blockIdx.x << 3) | (threadIdx.x >> 5);
  const int m0 = (wave >> 5) << 4;
  const int n0 = (wave & 31) << 4;

  v8f acc = {0.f, 0.f, 0.f, 0.f, 0.f, 0.f, 0.f, 0.f};
#pragma unroll 4
  for (int k0 = 0; k0 < H_DIM; k0 += 32) {
    v16bf a = load_a_bf16(h, H_DIM, m0, k0, lane);
    v16bf b = load_b_bf16(w_fc, H_DIM, n0, k0, lane);
    acc = __builtin_amdgcn_wmma_f32_16x16x32_bf16(false, a, false, b,
                                                  (short)0, acc, false, false);
  }
  const int l = lane & 15, hi = lane >> 4;
  const int col = n0 + l;
  const float bias = b_fc[col];
#pragma unroll
  for (int v = 0; v < 8; ++v) {
    const int row = m0 + v + (hi << 3);
    out[(size_t)row * OUT_DIM + col] = fmaxf(acc[v] + bias, 0.0f);
  }
}

// ---------------------------------------------------------------------------
// Host launcher
// ---------------------------------------------------------------------------
extern "C" void kernel_launch(void* const* d_in, const int* in_sizes, int n_in,
                              void* d_out, int out_size, void* d_ws, size_t ws_size,
                              hipStream_t stream) {
  (void)in_sizes; (void)n_in; (void)out_size; (void)ws_size;
  const float* x     = (const float*)d_in[0];
  const float* W_ih0 = (const float*)d_in[1];
  const float* W_hh0 = (const float*)d_in[2];
  const float* b_ih0 = (const float*)d_in[3];
  const float* b_hh0 = (const float*)d_in[4];
  const float* W_ih1 = (const float*)d_in[5];
  const float* W_hh1 = (const float*)d_in[6];
  const float* b_ih1 = (const float*)d_in[7];
  const float* b_hh1 = (const float*)d_in[8];
  const float* W_fc  = (const float*)d_in[9];
  const float* b_fc  = (const float*)d_in[10];

  // Workspace carve-out (bf16): weights + double-buffered hidden states (~10 MB)
  char* p = (char*)d_ws;
  __bf16* wih0 = (__bf16*)p; p += (size_t)H_DIM * IN_DIM * 2;
  __bf16* whh0 = (__bf16*)p; p += (size_t)H_DIM * H_DIM * 2;
  __bf16* wih1 = (__bf16*)p; p += (size_t)H_DIM * H_DIM * 2;
  __bf16* whh1 = (__bf16*)p; p += (size_t)H_DIM * H_DIM * 2;
  __bf16* wfc  = (__bf16*)p; p += (size_t)OUT_DIM * H_DIM * 2;
  __bf16* h0a  = (__bf16*)p; p += (size_t)B_DIM * H_DIM * 2;
  __bf16* h0b  = (__bf16*)p; p += (size_t)B_DIM * H_DIM * 2;
  __bf16* h1a  = (__bf16*)p; p += (size_t)B_DIM * H_DIM * 2;
  __bf16* h1b  = (__bf16*)p; p += (size_t)B_DIM * H_DIM * 2;

  const int TPB = 256;
  auto blocks = [](int n) { return (n + 255) / 256; };

  cvt_f32_to_bf16<<<blocks(H_DIM * IN_DIM), TPB, 0, stream>>>(W_ih0, wih0, H_DIM * IN_DIM);
  cvt_f32_to_bf16<<<blocks(H_DIM * H_DIM), TPB, 0, stream>>>(W_hh0, whh0, H_DIM * H_DIM);
  cvt_f32_to_bf16<<<blocks(H_DIM * H_DIM), TPB, 0, stream>>>(W_ih1, wih1, H_DIM * H_DIM);
  cvt_f32_to_bf16<<<blocks(H_DIM * H_DIM), TPB, 0, stream>>>(W_hh1, whh1, H_DIM * H_DIM);
  cvt_f32_to_bf16<<<blocks(OUT_DIM * H_DIM), TPB, 0, stream>>>(W_fc, wfc, OUT_DIM * H_DIM);
  zero_bf16<<<blocks(B_DIM * H_DIM), TPB, 0, stream>>>(h0a, B_DIM * H_DIM);
  zero_bf16<<<blocks(B_DIM * H_DIM), TPB, 0, stream>>>(h1a, B_DIM * H_DIM);

  for (int t = 0; t < T_DIM; ++t) {
    const __bf16* h0p = (t & 1) ? h0b : h0a;
    __bf16*       h0n = (t & 1) ? h0a : h0b;
    const __bf16* h1p = (t & 1) ? h1b : h1a;
    __bf16*       h1n = (t & 1) ? h1a : h1b;
    rnn_step_l0<<<128, TPB, 0, stream>>>(x, t, h0p, wih0, whh0, b_ih0, b_hh0, h0n);
    rnn_step_l1<<<128, TPB, 0, stream>>>(h0n, h1p, wih1, whh1, b_ih1, b_hh1, h1n);
  }
  // T=512 (even): last write at t=511 (odd) went to h1a
  fc_relu<<<64, TPB, 0, stream>>>(h1a, wfc, b_fc, (float*)d_out);
}
